// TeacherFocingLSTM_12867722019350
// MI455X (gfx1250) — matrix operand order repaired and checked by reading, compiled, not run on hardware
//
#include <hip/hip_runtime.h>
#include <hip/hip_bf16.h>

typedef __attribute__((ext_vector_type(8)))  __bf16 v8bf;
typedef __attribute__((ext_vector_type(16))) __bf16 v16bf;
typedef __attribute__((ext_vector_type(8)))  float  v8f;

#define BATCH 512
#define DIM   64
#define HID   512
#define THIST 256
#define FUT   32
#define TTOT  (THIST + FUT)
#define GATES 2048 /* 4*HID */

__device__ __forceinline__ float sigmoidf_(float x) { return 1.0f / (1.0f + __expf(-x)); }

// A-fragment (16x32 bf16): lane holds row M=lane%16; two contiguous 8-elem k-chunks
// at k0 + 16*(lane/16) and k0 + 16*(lane/16) + 8. p already includes row*lda + k0 + koff.
__device__ __forceinline__ v16bf load_a_frag(const __bf16* p) {
  v8bf lo = *(const v8bf*)(p);
  v8bf hi = *(const v8bf*)(p + 8);
  return __builtin_shufflevector(lo, hi, 0,1,2,3,4,5,6,7,8,9,10,11,12,13,14,15);
}

#define WMMA_BF16(A, B, C) \
  __builtin_amdgcn_wmma_f32_16x16x32_bf16(false, (A), false, (B), (short)0, (C), false, false)

// 3-term hi/lo product: C += Ah*Bl + Al*Bh + Ah*Bh  (lo*lo dropped, ~2^-16 rel err)
__device__ __forceinline__ v8f mma3(v16bf ah, v16bf al, v16bf bh, v16bf bl, v8f c) {
  c = WMMA_BF16(al, bh, c);
  c = WMMA_BF16(ah, bl, c);
  c = WMMA_BF16(ah, bh, c);
  return c;
}

__device__ __forceinline__ void split_store(__bf16* hi, __bf16* lo, size_t idx, float v) {
  __bf16 h = (__bf16)v;
  hi[idx] = h;
  lo[idx] = (__bf16)(v - (float)h);
}

// ---------------------------------------------------------------------------
// Fused LSTM layer step: gates = A0*W0^T + A1*W1^T + bias; cell update fused.
// Block = 128 threads (4 waves). Wave w computes gate w's 32x32 tile of the
// block's (m0,n0) region: 2x2 WMMA C-tiles, K = 512 + 512, hi/lo bf16 split.
// Gate exchange through LDS, then elementwise sigmoid/tanh cell update.
// ---------------------------------------------------------------------------
__global__ void __launch_bounds__(128) lstm_layer_fused(
    const __bf16* __restrict__ A0h, const __bf16* __restrict__ A0l, // BATCH x HID
    const __bf16* __restrict__ W0h, const __bf16* __restrict__ W0l, // GATES x HID
    const __bf16* __restrict__ A1h, const __bf16* __restrict__ A1l, // BATCH x HID (h_prev)
    const __bf16* __restrict__ W1h, const __bf16* __restrict__ W1l, // GATES x HID
    const float*  __restrict__ bias,  // GATES (b_ih + b_hh)
    float*        __restrict__ c_io,  // BATCH x HID (in-place)
    __bf16*       __restrict__ hoh,   // h_out hi
    __bf16*       __restrict__ hol)   // h_out lo
{
  __shared__ float smem[4 * 32 * 32];
  const int tid  = threadIdx.x;
  const int lane = tid & 31;
  const int gate = tid >> 5;                 // 0:i 1:f 2:g 3:o
  const int m0   = (blockIdx.x >> 4) * 32;
  const int n0   = (blockIdx.x & 15) * 32;

  const int rA   = lane & 15;
  const int koff = (lane >> 4) * 16;
  const int kb16 = (lane >> 4) * 16;
  const int nG   = gate * HID + n0;          // column base inside GATES

  v8f acc00 = {}, acc01 = {}, acc10 = {}, acc11 = {};

  for (int seg = 0; seg < 2; ++seg) {
    const __bf16* Ah = seg ? A1h : A0h;
    const __bf16* Al = seg ? A1l : A0l;
    const __bf16* Wh = seg ? W1h : W0h;
    const __bf16* Wl = seg ? W1l : W0l;
    const size_t aoff0 = (size_t)(m0 + rA) * HID + koff;
    const size_t aoff1 = (size_t)(m0 + 16 + rA) * HID + koff;
    const size_t boff0 = (size_t)(nG + rA) * HID + kb16;
    const size_t boff1 = (size_t)(nG + 16 + rA) * HID + kb16;
    #pragma unroll 2
    for (int k0 = 0; k0 < HID; k0 += 32) {
      v16bf a0h = load_a_frag(Ah + aoff0 + k0);
      v16bf a0l = load_a_frag(Al + aoff0 + k0);
      v16bf a1h = load_a_frag(Ah + aoff1 + k0);
      v16bf a1l = load_a_frag(Al + aoff1 + k0);
      v16bf b0h = *(const v16bf*)(Wh + boff0 + k0);
      v16bf b0l = *(const v16bf*)(Wl + boff0 + k0);
      v16bf b1h = *(const v16bf*)(Wh + boff1 + k0);
      v16bf b1l = *(const v16bf*)(Wl + boff1 + k0);
      acc00 = mma3(a0h, a0l, b0h, b0l, acc00);
      acc01 = mma3(a0h, a0l, b1h, b1l, acc01);
      acc10 = mma3(a1h, a1l, b0h, b0l, acc10);
      acc11 = mma3(a1h, a1l, b1h, b1l, acc11);
    }
  }

  // Epilogue: C layout => element r sits at (M = r + 8*(lane/16), N = lane%16).
  float* gl = smem + gate * 1024;
  const int mhalf = 8 * (lane >> 4);
  const float bv0 = bias[gate * HID + n0 + rA];
  const float bv1 = bias[gate * HID + n0 + 16 + rA];
  #pragma unroll
  for (int r = 0; r < 8; ++r) {
    const int mA = mhalf + r;
    gl[(mA)      * 32 + rA]      = acc00[r] + bv0;
    gl[(mA)      * 32 + 16 + rA] = acc01[r] + bv1;
    gl[(mA + 16) * 32 + rA]      = acc10[r] + bv0;
    gl[(mA + 16) * 32 + 16 + rA] = acc11[r] + bv1;
  }
  __syncthreads();

  #pragma unroll
  for (int i = 0; i < 8; ++i) {
    const int idx = tid + i * 128;           // 0..1023 over the 32x32 tile
    const float gi = smem[idx];
    const float gf = smem[1024 + idx];
    const float gg = smem[2048 + idx];
    const float go = smem[3072 + idx];
    const int ml = idx >> 5, nl = idx & 31;
    const size_t g = (size_t)(m0 + ml) * HID + (n0 + nl);
    const float cn = sigmoidf_(gf) * c_io[g] + sigmoidf_(gi) * tanhf(gg);
    c_io[g] = cn;
    split_store(hoh, hol, g, sigmoidf_(go) * tanhf(cn));
  }
}

// ---------------------------------------------------------------------------
// z = X * W_in^T + b_in   (M=512, N=512, K=64) -> hi/lo bf16
// One 16x16 tile per wave; 256 blocks x 4 waves = 1024 tiles.
// ---------------------------------------------------------------------------
__global__ void __launch_bounds__(128) z_gemm(
    const __bf16* __restrict__ Xh, const __bf16* __restrict__ Xl, int ldx,
    const __bf16* __restrict__ Wh, const __bf16* __restrict__ Wl, // HID x DIM
    const float*  __restrict__ b_in,
    __bf16*       __restrict__ Zh, __bf16* __restrict__ Zl)       // BATCH x HID
{
  const int tid  = threadIdx.x;
  const int lane = tid & 31;
  const int wid  = blockIdx.x * 4 + (tid >> 5); // 0..1023
  const int m0   = (wid >> 5) * 16;
  const int n0   = (wid & 31) * 16;
  const int rA   = lane & 15;
  const int koff = (lane >> 4) * 16;
  const int kb16 = (lane >> 4) * 16;

  v8f acc = {};
  const size_t aoff = (size_t)(m0 + rA) * ldx + koff;
  const size_t boff = (size_t)(n0 + rA) * DIM + kb16;
  #pragma unroll
  for (int k0 = 0; k0 < DIM; k0 += 32) {
    v16bf ah = load_a_frag(Xh + aoff + k0);
    v16bf al = load_a_frag(Xl + aoff + k0);
    v16bf bh = *(const v16bf*)(Wh + boff + k0);
    v16bf bl = *(const v16bf*)(Wl + boff + k0);
    acc = mma3(ah, al, bh, bl, acc);
  }
  const int mhalf = 8 * (lane >> 4);
  const float bv = b_in[n0 + rA];
  #pragma unroll
  for (int r = 0; r < 8; ++r) {
    const size_t g = (size_t)(m0 + mhalf + r) * HID + n0 + rA;
    split_store(Zh, Zl, g, acc[r] + bv);
  }
}

// ---------------------------------------------------------------------------
// out = H * W_out^T + b_out  (M=512, N=64, K=512)
// writes f32 slice of d_out (stride TTOT*DIM) and hi/lo bf16 copy for feedback.
// 32 blocks x 4 waves = 128 tiles (32 M-tiles x 4 N-tiles).
// ---------------------------------------------------------------------------
__global__ void __launch_bounds__(128) out_gemm(
    const __bf16* __restrict__ Hh, const __bf16* __restrict__ Hl,  // BATCH x HID
    const __bf16* __restrict__ Wh, const __bf16* __restrict__ Wl,  // DIM x HID
    const float*  __restrict__ b_out,
    float*        __restrict__ outp,   // d_out + t*DIM, row stride TTOT*DIM
    __bf16*       __restrict__ obh, __bf16* __restrict__ obl)      // BATCH x DIM
{
  const int tid  = threadIdx.x;
  const int lane = tid & 31;
  const int wid  = blockIdx.x * 4 + (tid >> 5); // 0..127
  const int m0   = (wid >> 2) * 16;
  const int n0   = (wid & 3) * 16;
  const int rA   = lane & 15;
  const int koff = (lane >> 4) * 16;
  const int kb16 = (lane >> 4) * 16;

  v8f acc = {};
  const size_t aoff = (size_t)(m0 + rA) * HID + koff;
  const size_t boff = (size_t)(n0 + rA) * HID + kb16;
  #pragma unroll 4
  for (int k0 = 0; k0 < HID; k0 += 32) {
    v16bf ah = load_a_frag(Hh + aoff + k0);
    v16bf al = load_a_frag(Hl + aoff + k0);
    v16bf bh = *(const v16bf*)(Wh + boff + k0);
    v16bf bl = *(const v16bf*)(Wl + boff + k0);
    acc = mma3(ah, al, bh, bl, acc);
  }
  const int mhalf = 8 * (lane >> 4);
  const float bv = b_out[n0 + rA];
  #pragma unroll
  for (int r = 0; r < 8; ++r) {
    const int gm = m0 + mhalf + r;
    const float v = acc[r] + bv;
    outp[(size_t)gm * (TTOT * DIM) + n0 + rA] = v;
    split_store(obh, obl, (size_t)gm * DIM + n0 + rA, v);
  }
}

// ------------------------- utility kernels ---------------------------------
__global__ void split_f32_bf16(const float* __restrict__ s,
                               __bf16* __restrict__ hi, __bf16* __restrict__ lo, int n) {
  for (int i = blockIdx.x * blockDim.x + threadIdx.x; i < n; i += gridDim.x * blockDim.x) {
    float x = s[i];
    __bf16 h = (__bf16)x;
    hi[i] = h;
    lo[i] = (__bf16)(x - (float)h);
  }
}
__global__ void bias_sum_k(const float* __restrict__ a, const float* __restrict__ b,
                           float* __restrict__ o, int n) {
  for (int i = blockIdx.x * blockDim.x + threadIdx.x; i < n; i += gridDim.x * blockDim.x)
    o[i] = a[i] + b[i];
}
__global__ void zero_f32_k(float* p, int n) {
  for (int i = blockIdx.x * blockDim.x + threadIdx.x; i < n; i += gridDim.x * blockDim.x)
    p[i] = 0.0f;
}
__global__ void zero_bf16_k(__bf16* p, int n) {
  for (int i = blockIdx.x * blockDim.x + threadIdx.x; i < n; i += gridDim.x * blockDim.x)
    p[i] = (__bf16)0.0f;
}

// ---------------------------------------------------------------------------
extern "C" void kernel_launch(void* const* d_in, const int* in_sizes, int n_in,
                              void* d_out, int out_size, void* d_ws, size_t ws_size,
                              hipStream_t stream) {
  const float* x_hist = (const float*)d_in[0];
  const float* W_in   = (const float*)d_in[1];
  const float* b_in   = (const float*)d_in[2];
  const float* W_ih   = (const float*)d_in[3];
  const float* W_hh   = (const float*)d_in[4];
  const float* b_ih   = (const float*)d_in[5];
  const float* b_hh   = (const float*)d_in[6];
  const float* W_out  = (const float*)d_in[7];
  const float* b_out  = (const float*)d_in[8];
  (void)in_sizes; (void)n_in; (void)out_size; (void)ws_size;

  // workspace carve (256B aligned chunks)
  char* w = (char*)d_ws;
  auto carve = [&](size_t bytes) -> void* {
    void* p = (void*)w;
    w += (bytes + 255) & ~(size_t)255;
    return p;
  };
  const size_t WIH = (size_t)2 * GATES * HID;   // per hi/lo array elems
  __bf16* winh  = (__bf16*)carve((size_t)HID * DIM * 2);
  __bf16* winl  = (__bf16*)carve((size_t)HID * DIM * 2);
  __bf16* wihh  = (__bf16*)carve(WIH * 2);
  __bf16* wihl  = (__bf16*)carve(WIH * 2);
  __bf16* whhh  = (__bf16*)carve(WIH * 2);
  __bf16* whhl  = (__bf16*)carve(WIH * 2);
  __bf16* wouth = (__bf16*)carve((size_t)DIM * HID * 2);
  __bf16* woutl = (__bf16*)carve((size_t)DIM * HID * 2);
  __bf16* xh    = (__bf16*)carve((size_t)BATCH * THIST * DIM * 2);
  __bf16* xl    = (__bf16*)carve((size_t)BATCH * THIST * DIM * 2);
  float*  bias01= (float*) carve((size_t)2 * GATES * 4);
  __bf16* zh    = (__bf16*)carve((size_t)BATCH * HID * 2);
  __bf16* zl    = (__bf16*)carve((size_t)BATCH * HID * 2);
  __bf16* h0h[2], *h0l[2], *h1h[2], *h1l[2];
  for (int i = 0; i < 2; ++i) {
    h0h[i] = (__bf16*)carve((size_t)BATCH * HID * 2);
    h0l[i] = (__bf16*)carve((size_t)BATCH * HID * 2);
    h1h[i] = (__bf16*)carve((size_t)BATCH * HID * 2);
    h1l[i] = (__bf16*)carve((size_t)BATCH * HID * 2);
  }
  float*  c0   = (float*) carve((size_t)BATCH * HID * 4);
  float*  c1   = (float*) carve((size_t)BATCH * HID * 4);
  __bf16* obh  = (__bf16*)carve((size_t)BATCH * DIM * 2);
  __bf16* obl  = (__bf16*)carve((size_t)BATCH * DIM * 2);

  // prep: split weights/activations to hi/lo bf16, fold biases, zero state
  split_f32_bf16<<<64,  256, 0, stream>>>(W_in,  winh,  winl,  HID * DIM);
  split_f32_bf16<<<1024,256, 0, stream>>>(W_ih,  wihh,  wihl,  (int)WIH);
  split_f32_bf16<<<1024,256, 0, stream>>>(W_hh,  whhh,  whhl,  (int)WIH);
  split_f32_bf16<<<64,  256, 0, stream>>>(W_out, wouth, woutl, DIM * HID);
  split_f32_bf16<<<1024,256, 0, stream>>>(x_hist, xh,   xl,    BATCH * THIST * DIM);
  bias_sum_k    <<<16,  256, 0, stream>>>(b_ih, b_hh, bias01, 2 * GATES);
  for (int i = 0; i < 2; ++i) {
    zero_bf16_k<<<256, 256, 0, stream>>>(h0h[i], BATCH * HID);
    zero_bf16_k<<<256, 256, 0, stream>>>(h0l[i], BATCH * HID);
    zero_bf16_k<<<256, 256, 0, stream>>>(h1h[i], BATCH * HID);
    zero_bf16_k<<<256, 256, 0, stream>>>(h1l[i], BATCH * HID);
  }
  zero_f32_k<<<256, 256, 0, stream>>>(c0, BATCH * HID);
  zero_f32_k<<<256, 256, 0, stream>>>(c1, BATCH * HID);

  int p = 0;
  for (int t = 0; t < TTOT; ++t) {
    if (t < THIST)
      z_gemm<<<256, 128, 0, stream>>>(xh + (size_t)t * DIM, xl + (size_t)t * DIM,
                                      THIST * DIM, winh, winl, b_in, zh, zl);
    else
      z_gemm<<<256, 128, 0, stream>>>(obh, obl, DIM, winh, winl, b_in, zh, zl);

    lstm_layer_fused<<<256, 128, 0, stream>>>(
        zh, zl, wihh, wihl, h0h[p], h0l[p], whhh, whhl,
        bias01, c0, h0h[1 - p], h0l[1 - p]);
    lstm_layer_fused<<<256, 128, 0, stream>>>(
        h0h[1 - p], h0l[1 - p], wihh + WIH / 2, wihl + WIH / 2,
        h1h[p], h1l[p], whhh + WIH / 2, whhl + WIH / 2,
        bias01 + GATES, c1, h1h[1 - p], h1l[1 - p]);

    out_gemm<<<32, 128, 0, stream>>>(
        h1h[1 - p], h1l[1 - p], wouth, woutl, b_out,
        (float*)d_out + (size_t)t * DIM, obh, obl);
    p ^= 1;
  }
}
